// Discriminiator_32384053412272
// MI455X (gfx1250) — matrix operand order repaired and checked by reading, not compiled, not run on hardware
//
#include <hip/hip_runtime.h>
#include <hip/hip_bf16.h>

typedef __attribute__((ext_vector_type(16))) __bf16 v16bf;
typedef __attribute__((ext_vector_type(8)))  float  v8f;
typedef __attribute__((ext_vector_type(4)))  int    v4i;

#define BB 4096
#define TT 256
#define DD 128
#define HH 32
#define XSTRIDE 130      // padded row stride (floats) for x staging
#define HSTRIDE 34       // padded row stride (floats) for h staging

#if defined(__has_builtin)
#if __has_builtin(__builtin_amdgcn_global_load_async_to_lds_b128)
#define HAVE_ASYNC_LDS 1
#endif
#endif

#ifdef HAVE_ASYNC_LDS
typedef __attribute__((address_space(1))) v4i gv4i_t;   // global v4i
typedef __attribute__((address_space(3))) v4i lv4i_t;   // LDS v4i
__device__ __forceinline__ void async_b128(const void* src, void* lds) {
    __builtin_amdgcn_global_load_async_to_lds_b128(
        (gv4i_t*)(v4i*)const_cast<void*>(src), (lv4i_t*)(v4i*)lds, 0, 0);
}
__device__ __forceinline__ void wait_async_le16() {
#if __has_builtin(__builtin_amdgcn_s_wait_asynccnt)
    __builtin_amdgcn_s_wait_asynccnt(16);
#else
    asm volatile("s_wait_asynccnt 0x10" ::: "memory");
#endif
}
__device__ __forceinline__ void wait_async_0() {
#if __has_builtin(__builtin_amdgcn_s_wait_asynccnt)
    __builtin_amdgcn_s_wait_asynccnt(0);
#else
    asm volatile("s_wait_asynccnt 0x0" ::: "memory");
#endif
}
#endif

__device__ __forceinline__ unsigned bf16bits(float f) {
    unsigned u = __float_as_uint(f);
    unsigned r = (u + 0x7FFFu + ((u >> 16) & 1u)) >> 16;   // RNE
    return r & 0xFFFFu;
}

__device__ __forceinline__ float fast_sigmoid(float x) {
    return __builtin_amdgcn_rcpf(1.0f + __builtin_amdgcn_exp2f(-1.44269504088896f * x));
}

__device__ __forceinline__ float fast_tanh(float x) {
    float e = __builtin_amdgcn_exp2f(2.88539008177793f * x);   // exp(2x)
    return (e - 1.0f) * __builtin_amdgcn_rcpf(e + 1.0f);
}

// One wave (32 lanes) = one workgroup owns 16 batch rows for the whole sequence.
// gates[16x128] = [x_t(16x128) | h(16x32)] @ [W_ih | W_hh]^T  via 5x8 bf16 WMMAs.
__global__ void __launch_bounds__(32)
lstm_fused_kernel(const float* __restrict__ feed,
                  const float* __restrict__ W_ih,
                  const float* __restrict__ W_hh,
                  const float* __restrict__ b_ih,
                  const float* __restrict__ b_hh,
                  const float* __restrict__ W_out,
                  const float* __restrict__ b_out,
                  const float* __restrict__ h0,
                  const float* __restrict__ c0,
                  float* __restrict__ out) {
    // B-operand pack: 40 tiles (5 K-chunks x 8 N-tiles), 32 lanes x 32B each.
    __shared__ __align__(16) unsigned bpack[40 * 32 * 8];          // 40 KB
    __shared__ __align__(16) float    xbuf[2][16 * XSTRIDE];       // 2 x 8.3 KB (double buffer)
    __shared__ __align__(16) float    hbuf[16 * HSTRIDE];          // 2.2 KB

    const int ln   = threadIdx.x & 31;
    const int ln15 = ln & 15;
    const int hi8  = (ln < 16) ? 0 : 8;    // C/D row offset for upper half-wave
    const int aoff = (ln < 16) ? 0 : 8;    // A-layout K offset for upper half-wave
    const int b0   = blockIdx.x * 16;

    // ---- Pack combined weight matrix [D+H=160, G=128] into WMMA B layout (bf16) ----
    // tile = kc*8+nt ; lane L: N = nt*16 + (L&15), K = kc*32 + ((L&16)?16:0) + 0..15
    for (int s = threadIdx.x; s < 40 * 32; s += 32) {
        const int tile = s >> 5, L = s & 31;
        const int kc = tile >> 3, nt = tile & 7;
        const int N  = nt * 16 + (L & 15);
        const int k0 = kc * 32 + ((L & 16) ? 16 : 0);
        unsigned* dst = &bpack[tile * 256 + L * 8];
        #pragma unroll
        for (int w2 = 0; w2 < 8; ++w2) {
            const int k = k0 + 2 * w2;
            float f0, f1;
            if (k < DD) { f0 = W_ih[N * DD + k];        f1 = W_ih[N * DD + k + 1]; }
            else        { f0 = W_hh[N * HH + (k - DD)]; f1 = W_hh[N * HH + (k - DD) + 1]; }
            dst[w2] = bf16bits(f0) | (bf16bits(f1) << 16);
        }
    }
    __syncthreads();
    const v16bf* bp = reinterpret_cast<const v16bf*>(bpack);

    // ---- Per-lane gate bias (b_ih + b_hh), broadcast over C rows ----
    float biasv[8];
    #pragma unroll
    for (int nt = 0; nt < 8; ++nt)
        biasv[nt] = b_ih[nt * 16 + ln15] + b_hh[nt * 16 + ln15];

    // ---- h/c state in C/D accumulator layout: [j=H-tile][v=row] ----
    float hS[2][8], cS[2][8];
    #pragma unroll
    for (int j = 0; j < 2; ++j)
        #pragma unroll
        for (int v = 0; v < 8; ++v) {
            const int r = b0 + v + hi8;
            const int cix = j * 16 + ln15;
            hS[j][v] = h0[r * HH + cix];
            cS[j][v] = c0[r * HH + cix];
        }

#ifdef HAVE_ASYNC_LDS
    // Prime the pipeline: async-stage x_0 into buffer 0.
    {
        const float* f0p = feed + (size_t)b0 * TT * DD;
        #pragma unroll
        for (int i = 0; i < 16; ++i)
            async_b128(f0p + (size_t)i * TT * DD + ln * 4, &xbuf[0][i * XSTRIDE + ln * 4]);
    }
#endif

    #pragma unroll 1
    for (int t = 0; t < TT; ++t) {
        float* xw = xbuf[t & 1];
#ifdef HAVE_ASYNC_LDS
        // ---- Issue async staging of x_{t+1} into the other buffer ----
        {
            const int tn = (t + 1 < TT) ? (t + 1) : (TT - 1);
            const float* fnp = feed + ((size_t)b0 * TT + tn) * DD;
            float* xn = xbuf[(t + 1) & 1];
            #pragma unroll
            for (int i = 0; i < 16; ++i) {
                const float* src = fnp + (size_t)i * TT * DD + ln * 4;
                async_b128(src, &xn[i * XSTRIDE + ln * 4]);
                __builtin_prefetch(src + DD, 0, 0);   // warm L2 for t+2
            }
        }
        wait_async_le16();   // x_t's 16 transfers have landed in LDS
#else
        // ---- Synchronous staging fallback (coalesced 512B per row) ----
        {
            const float* ftbase = feed + ((size_t)b0 * TT + t) * DD;
            #pragma unroll
            for (int i = 0; i < 16; ++i) {
                const float* src = ftbase + (size_t)i * TT * DD + ln * 4;
                *reinterpret_cast<float4*>(&xw[i * XSTRIDE + ln * 4]) =
                    *reinterpret_cast<const float4*>(src);
                __builtin_prefetch(src + DD, 0, 0);
            }
        }
#endif
        // ---- Stage h (C-layout regs -> row-major LDS) for A-layout regather ----
        #pragma unroll
        for (int j = 0; j < 2; ++j)
            #pragma unroll
            for (int v = 0; v < 8; ++v)
                hbuf[(v + hi8) * HSTRIDE + j * 16 + ln15] = hS[j][v];
        asm volatile("" ::: "memory");

        // ---- Gather A operands: 4 chunks from x, 1 chunk from h ----
        v16bf A[5];
        #pragma unroll
        for (int kc = 0; kc < 4; ++kc)
            #pragma unroll
            for (int v = 0; v < 8; ++v) {
                const int d = kc * 32 + 2 * aoff + 2 * v;
                const float2 f2 = *reinterpret_cast<const float2*>(&xw[ln15 * XSTRIDE + d]);
                A[kc][2 * v]     = (__bf16)f2.x;
                A[kc][2 * v + 1] = (__bf16)f2.y;
            }
        #pragma unroll
        for (int v = 0; v < 8; ++v) {
            const int d = 2 * aoff + 2 * v;
            const float2 f2 = *reinterpret_cast<const float2*>(&hbuf[ln15 * HSTRIDE + d]);
            A[4][2 * v]     = (__bf16)f2.x;
            A[4][2 * v + 1] = (__bf16)f2.y;
        }

        // ---- 8 gate tiles, K=160 in 5 WMMA steps each ----
        v8f C[8];
        #pragma unroll
        for (int nt = 0; nt < 8; ++nt) {
            v8f c;
            #pragma unroll
            for (int v = 0; v < 8; ++v) c[v] = biasv[nt];
            C[nt] = c;
        }
        #pragma unroll
        for (int kc = 0; kc < 5; ++kc)
            #pragma unroll
            for (int nt = 0; nt < 8; ++nt) {
                const v16bf bt = bp[(kc * 8 + nt) * 32 + ln];
                C[nt] = __builtin_amdgcn_wmma_f32_16x16x32_bf16(
                    false, A[kc], false, bt, (short)0, C[nt], false, false);
            }

        // ---- Elementwise LSTM cell (all operands share the C/D layout) ----
        #pragma unroll
        for (int j = 0; j < 2; ++j)
            #pragma unroll
            for (int v = 0; v < 8; ++v) {
                const float ig = fast_sigmoid(C[0 + j][v]);
                const float fg = fast_sigmoid(C[2 + j][v]);
                const float gg = fast_tanh   (C[4 + j][v]);
                const float og = fast_sigmoid(C[6 + j][v]);
                const float cv = fg * cS[j][v] + ig * gg;
                cS[j][v] = cv;
                hS[j][v] = og * fast_tanh(cv);
            }
        asm volatile("" ::: "memory");
    }

#ifdef HAVE_ASYNC_LDS
    wait_async_0();   // drain the (redundant) final prefetch transfers
#endif

    // ---- Final linear: out[b,0:2] = h_last @ W_out.T + b_out ----
    #pragma unroll
    for (int j = 0; j < 2; ++j)
        #pragma unroll
        for (int v = 0; v < 8; ++v)
            hbuf[(v + hi8) * HSTRIDE + j * 16 + ln15] = hS[j][v];
    asm volatile("" ::: "memory");
    __syncthreads();

    if (ln < 16) {
        float a0 = b_out[0], a1 = b_out[1];
        #pragma unroll
        for (int k = 0; k < HH; ++k) {
            const float hv = hbuf[ln * HSTRIDE + k];
            a0 += hv * W_out[k];
            a1 += hv * W_out[HH + k];
        }
        out[(size_t)(b0 + ln) * 2 + 0] = a0;
        out[(size_t)(b0 + ln) * 2 + 1] = a1;
    }
}

extern "C" void kernel_launch(void* const* d_in, const int* in_sizes, int n_in,
                              void* d_out, int out_size, void* d_ws, size_t ws_size,
                              hipStream_t stream) {
    (void)in_sizes; (void)n_in; (void)out_size; (void)d_ws; (void)ws_size;
    const float* feed  = (const float*)d_in[0];
    const float* W_ih  = (const float*)d_in[1];
    const float* W_hh  = (const float*)d_in[2];
    const float* b_ih  = (const float*)d_in[3];
    const float* b_hh  = (const float*)d_in[4];
    const float* W_out = (const float*)d_in[5];
    const float* b_out = (const float*)d_in[6];
    const float* h0    = (const float*)d_in[7];
    const float* c0    = (const float*)d_in[8];
    float* out = (float*)d_out;

    // 4096 batch rows / 16 rows per wave-workgroup = 256 blocks of 32 threads
    lstm_fused_kernel<<<dim3(BB / 16), dim3(32), 0, stream>>>(
        feed, W_ih, W_hh, b_ih, b_hh, W_out, b_out, h0, c0, out);
}
